// TSPointConv_45981919871337
// MI455X (gfx1250) — compile-verified, matrix operand order checked by reading, not run on hardware
//
#include <hip/hip_runtime.h>
#include <hip/hip_bf16.h>

// ---------------------------------------------------------------------------
// MI455X (gfx1250) fused PointConv-style GNN forward.
//  - bf16 WMMA (v_wmma_f32_16x16x32_bf16) for the heavy MLPs, f32 accumulate
//  - local block fully fused: LDS-resident activations, GN via lane shuffles,
//    scatter-add into L2-resident agg buffer with float atomics
//  - wave = one GroupNorm group => GN needs only 16-lane butterflies
//  - multi-M-tile workgroups: each B fragment load feeds MT WMMAs, so the
//    weight stream from L2 is amortized 4x (local) / 2x (glob)
// ---------------------------------------------------------------------------

typedef unsigned short ushort_t;
typedef __attribute__((ext_vector_type(16))) __bf16        v16bf;
typedef __attribute__((ext_vector_type(8)))  float         v8f;
typedef __attribute__((ext_vector_type(4)))  unsigned int  uint4v;

#define EPS_GN 1e-5f

__device__ __forceinline__ unsigned short f2bf(float f) {
    unsigned u = __builtin_bit_cast(unsigned, f);
    u += 0x7FFFu + ((u >> 16) & 1u);            // round-to-nearest-even
    return (unsigned short)(u >> 16);
}
__device__ __forceinline__ float bf2f(unsigned short h) {
    return __builtin_bit_cast(float, ((unsigned)h) << 16);
}

// ---------------------------------------------------------------------------
// WMMA fragment helpers (wave32).
// A (16x32 bf16): lane = M (0..15, both halves), per lane two contiguous
//   8-element K chunks: [half*8 .. half*8+7] and [16+half*8 .. 16+half*8+7].
// B is pre-packed so each lane reads 16 contiguous bf16 (32B) per tile.
// C/D (16x16 f32): vgpr r, lane l -> M = r + (l>=16)*8, N = l&15.
// ---------------------------------------------------------------------------
__device__ __forceinline__ v16bf load_frag_A(const ushort_t* act, int kp,
                                             int ktile, int lane) {
    const int m = lane & 15;
    const int half = (lane >> 4) & 1;
    const ushort_t* p = act + (size_t)m * kp + ktile * 32 + half * 8;
    union { uint4v u[2]; v16bf v; } r;
    r.u[0] = *(const uint4v*)(p);
    r.u[1] = *(const uint4v*)(p + 16);
    return r.v;
}

__device__ __forceinline__ v16bf load_frag_B(const ushort_t* g, int lane) {
    union { uint4v u[2]; v16bf v; } r;
    const ushort_t* p = g + lane * 16;
    r.u[0] = *(const uint4v*)(p);
    r.u[1] = *(const uint4v*)(p + 8);
    return r.v;
}

// One wave computes an MT x TPW grid of 16x16 output tiles over K = NKT*32.
// Each B fragment is reused across MT M-tiles (key to L2-BW efficiency).
template <int MT, int TPW, int NKT>
__device__ __forceinline__ void wave_gemm(v8f (&acc)[MT][TPW], const ushort_t* act,
                                          int kp, const ushort_t* pw, int ntbase,
                                          int lane) {
#pragma unroll
    for (int mt = 0; mt < MT; ++mt)
#pragma unroll
        for (int t = 0; t < TPW; ++t) acc[mt][t] = {};
#pragma unroll 1
    for (int kt = 0; kt < NKT; ++kt) {
        v16bf a[MT];
#pragma unroll
        for (int mt = 0; mt < MT; ++mt)
            a[mt] = load_frag_A(act + (size_t)mt * 16 * kp, kp, kt, lane);
        if (kt + 1 < NKT)   // stream next B k-slab toward L2/L0
            __builtin_prefetch(pw + ((size_t)ntbase * NKT + kt + 1) * 512, 0, 1);
#pragma unroll
        for (int t = 0; t < TPW; ++t) {
            v16bf b = load_frag_B(pw + ((size_t)(ntbase + t) * NKT + kt) * 512, lane);
#pragma unroll
            for (int mt = 0; mt < MT; ++mt)
                acc[mt][t] = __builtin_amdgcn_wmma_f32_16x16x32_bf16(
                    false, a[mt], false, b, (short)0, acc[mt][t], false, false);
        }
    }
}

__device__ __forceinline__ float half16_sum(float v) {
    v += __shfl_xor(v, 1);
    v += __shfl_xor(v, 2);
    v += __shfl_xor(v, 4);
    v += __shfl_xor(v, 8);
    return v;
}

// GroupNorm (group == this wave's TPW*16 channels) + ReLU for one 16-row
// M-tile, store bf16 to LDS at sout (already offset to the M-tile's rows).
template <int TPW>
__device__ __forceinline__ void wave_gn_relu_store(v8f (&acc)[TPW], const float* gam,
                                                   const float* bet, int cbase,
                                                   ushort_t* sout, int cp, int lane) {
    const int half = (lane >> 4) & 1;
    float mu[8], rs[8];
    {
        float s[8], q[8];
#pragma unroll
        for (int r = 0; r < 8; ++r) { s[r] = 0.f; q[r] = 0.f; }
#pragma unroll
        for (int t = 0; t < TPW; ++t)
#pragma unroll
            for (int r = 0; r < 8; ++r) {
                float v = acc[t][r];
                s[r] += half16_sum(v);
                q[r] += half16_sum(v * v);
            }
        const float inv = 1.0f / (float)(TPW * 16);
#pragma unroll
        for (int r = 0; r < 8; ++r) {
            mu[r] = s[r] * inv;
            float var = q[r] * inv - mu[r] * mu[r];
            rs[r] = rsqrtf(var + EPS_GN);
        }
    }
#pragma unroll
    for (int t = 0; t < TPW; ++t) {
        const int c = cbase + t * 16 + (lane & 15);
        const float gm = gam[c], bt = bet[c];
#pragma unroll
        for (int r = 0; r < 8; ++r) {
            float y = (acc[t][r] - mu[r]) * rs[r] * gm + bt;
            y = fmaxf(y, 0.f);
            sout[(size_t)(r + half * 8) * cp + c] = f2bf(y);
        }
    }
}

// ---------------------------------------------------------------------------
// Weight packing: f32 [K,N] row-major -> bf16 WMMA-B fragment stream.
// Tile index = nt*NKT + kt; within tile: lane*16 + h with
//   n = 16*nt + (lane&15), k = 32*kt + 16*(h>>3) + 8*(lane>>4) + (h&7).
// ---------------------------------------------------------------------------
__global__ void pack_b_kernel(const float* __restrict__ W, ushort_t* __restrict__ out,
                              int Kreal, int Kpad, int Nout) {
    const int nkt = Kpad >> 5;
    const int total = Kpad * Nout;
    for (int idx = blockIdx.x * blockDim.x + threadIdx.x; idx < total;
         idx += gridDim.x * blockDim.x) {
        int elem = idx & 511, tile = idx >> 9;
        int nt = tile / nkt, kt = tile - nt * nkt;
        int lane = elem >> 4, h = elem & 15;
        int k = kt * 32 + ((h >> 3) << 4) + ((lane >> 4) << 3) + (h & 7);
        int n = nt * 16 + (lane & 15);
        float v = (k < Kreal) ? W[(size_t)k * Nout + n] : 0.f;
        out[idx] = f2bf(v);
    }
}

// ---------------------------------------------------------------------------
// Small per-node MLP blocks (enc / encpos): one thread per node, whole row in
// registers; weight indices are thread-uniform -> scalar loads. ~0.8 GFLOP.
// ---------------------------------------------------------------------------
template <int H, int NG>
__device__ __forceinline__ void gn_relu_reg(float* h, const float* g, const float* be) {
    const int GS = H / NG;
#pragma unroll
    for (int grp = 0; grp < NG; ++grp) {
        float mu = 0.f, qq = 0.f;
#pragma unroll
        for (int j = 0; j < GS; ++j) { float v = h[grp * GS + j]; mu += v; qq += v * v; }
        mu /= (float)GS;
        float rstd = rsqrtf(qq / (float)GS - mu * mu + EPS_GN);
#pragma unroll
        for (int j = 0; j < GS; ++j) {
            int c = grp * GS + j;
            h[c] = fmaxf((h[c] - mu) * rstd * g[c] + be[c], 0.f);
        }
    }
}

template <int DIN, int H, int NG>
__global__ void mlp_node_kernel(const float* __restrict__ x,
                                const float* w1, const float* b1, const float* g1,
                                const float* be1, const float* w2, const float* b2,
                                const float* g2, const float* be2, const float* w3,
                                const float* b3, ushort_t* __restrict__ feat,
                                int fstride, int coff, int n) {
    int i = blockIdx.x * blockDim.x + threadIdx.x;
    if (i >= n) return;
    float xin[DIN];
#pragma unroll
    for (int d = 0; d < DIN; ++d) xin[d] = x[(size_t)i * DIN + d];
    float h1[H], h2[H];
#pragma unroll 2
    for (int j = 0; j < H; ++j) {
        float a = b1[j];
#pragma unroll
        for (int k = 0; k < DIN; ++k) a += xin[k] * w1[k * H + j];
        h1[j] = a;
    }
    gn_relu_reg<H, NG>(h1, g1, be1);
#pragma unroll 2
    for (int j = 0; j < H; ++j) {
        float a = b2[j];
#pragma unroll 4
        for (int k = 0; k < H; ++k) a += h1[k] * w2[k * H + j];
        h2[j] = a;
    }
    gn_relu_reg<H, NG>(h2, g2, be2);
#pragma unroll 2
    for (int j = 0; j < H; ++j) {
        float a = b3[j];
#pragma unroll 4
        for (int k = 0; k < H; ++k) a += h2[k] * w3[k * H + j];
        feat[(size_t)i * fstride + coff + j] = f2bf(a);
    }
}

// ---------------------------------------------------------------------------
// Fused local message block: 64 messages per workgroup (4 M-tiles), 8 waves,
// wave = 64 output channels = one GN(8,512) group. Three bf16 WMMA GEMMs,
// activations in LDS, scatter-add into agg[N,512] via float atomics.
// ---------------------------------------------------------------------------
#define LROWS 64
#define LMT   4
__global__ __launch_bounds__(256) void local_kernel(
    const ushort_t* __restrict__ feat, const float* __restrict__ pos,
    const int* __restrict__ src, const int* __restrict__ dst,
    const ushort_t* pw1, const ushort_t* pw2, const ushort_t* pw3,
    const float* g1, const float* be1, const float* g2, const float* be2,
    const float* b3, float* __restrict__ agg, int nE, int nN) {
    __shared__ ushort_t sA[LROWS * 128];    // msg_in, K padded 100 -> 128 (16KB)
    __shared__ ushort_t sAct[LROWS * 512];  // inter-layer activations (64KB)
    __shared__ int sSrc[LROWS], sDst[LROWS], sValid[LROWS];

    const int tid = threadIdx.x;
    const int lane = tid & 31;
    const int wave = tid >> 5;
    const int rowBase = blockIdx.x * LROWS;

    if (tid < LROWS) {
        int g = rowBase + tid, s = 0, d = 0, v = 0;
        if (g < nE)            { s = src[g]; d = dst[g]; v = 1; }
        else if (g < nE + nN)  { s = d = g - nE; v = 1; }      // self loops
        sSrc[tid] = s; sDst[tid] = d; sValid[tid] = v;
    }
    __syncthreads();

    // Gather msg_in = [feat[src] (96 bf16) | pos[src]-pos[dst] (4) | pad]
    for (int idx = tid; idx < LROWS * 128; idx += 256) {
        int m = idx >> 7, k = idx & 127;
        ushort_t val = 0;
        if (sValid[m]) {
            int s = sSrc[m];
            if (k < 96)       val = feat[(size_t)s * 96 + k];
            else if (k < 100) val = f2bf(pos[(size_t)s * 4 + (k - 96)] -
                                         pos[(size_t)sDst[m] * 4 + (k - 96)]);
        }
        sA[idx] = val;
    }
    __syncthreads();

    v8f acc[LMT][4];
    // L1: [64,128] x [128,512]
    wave_gemm<LMT, 4, 4>(acc, sA, 128, pw1, wave * 4, lane);
#pragma unroll
    for (int mt = 0; mt < LMT; ++mt)
        wave_gn_relu_store<4>(acc[mt], g1, be1, wave * 64,
                              sAct + (size_t)mt * 16 * 512, 512, lane);
    __syncthreads();
    // L2: [64,512] x [512,512]
    wave_gemm<LMT, 4, 16>(acc, sAct, 512, pw2, wave * 4, lane);
    __syncthreads();  // all waves done reading sAct
#pragma unroll
    for (int mt = 0; mt < LMT; ++mt)
        wave_gn_relu_store<4>(acc[mt], g2, be2, wave * 64,
                              sAct + (size_t)mt * 16 * 512, 512, lane);
    __syncthreads();
    // L3: [64,512] x [512,512] (+b3), scatter-add to agg[dst]
    wave_gemm<LMT, 4, 16>(acc, sAct, 512, pw3, wave * 4, lane);

    const int half = (lane >> 4) & 1;
#pragma unroll
    for (int t = 0; t < 4; ++t) {
        const int c = wave * 64 + t * 16 + (lane & 15);
        const float b3c = b3[c];
#pragma unroll
        for (int mt = 0; mt < LMT; ++mt)
#pragma unroll
            for (int r = 0; r < 8; ++r) {
                int m = mt * 16 + r + half * 8;
                if (sValid[m])
                    atomicAdd(&agg[(size_t)sDst[m] * 512 + c], acc[mt][t][r] + b3c);
            }
    }
}

// ---------------------------------------------------------------------------
// Fused glob + gate blocks per 32-node tile (2 M-tiles). Wave = 32 channels =
// one GN(8,256) group. Emits h [N,256] f32 and gate [N] f32.
// ---------------------------------------------------------------------------
#define GROWS 32
#define GMT   2
__global__ __launch_bounds__(256) void glob_gate_kernel(
    const float* __restrict__ agg, const ushort_t* pg1, const ushort_t* pg2,
    const ushort_t* pg3, const float* gg1, const float* gb1, const float* gg2,
    const float* gb2, const float* gb3, const ushort_t* pt1, const ushort_t* pt2,
    const float* tg1, const float* tb1, const float* tg2, const float* tb2,
    const float* tw3, const float* tb3, float* __restrict__ hbuf,
    float* __restrict__ gateout, int nN) {
    __shared__ ushort_t sIn[GROWS * 512];   // 32KB
    __shared__ ushort_t sC[GROWS * 256];    // 16KB
    __shared__ float sRed[GROWS * 8];

    const int tid = threadIdx.x;
    const int lane = tid & 31;
    const int wave = tid >> 5;
    const int half = (lane >> 4) & 1;
    const int rowBase = blockIdx.x * GROWS;

    for (int idx = tid; idx < GROWS * 512; idx += 256) {
        int m = idx >> 9, k = idx & 511;
        int node = rowBase + m;
        sIn[idx] = f2bf(node < nN ? agg[(size_t)node * 512 + k] : 0.f);
    }
    __syncthreads();

    v8f acc[GMT][2];
    // glob: 512 -> 256 -> 256 -> 256
    wave_gemm<GMT, 2, 16>(acc, sIn, 512, pg1, wave * 2, lane);
#pragma unroll
    for (int mt = 0; mt < GMT; ++mt)
        wave_gn_relu_store<2>(acc[mt], gg1, gb1, wave * 32,
                              sC + (size_t)mt * 16 * 256, 256, lane);
    __syncthreads();
    wave_gemm<GMT, 2, 8>(acc, sC, 256, pg2, wave * 2, lane);
    __syncthreads();
#pragma unroll
    for (int mt = 0; mt < GMT; ++mt)
        wave_gn_relu_store<2>(acc[mt], gg2, gb2, wave * 32,
                              sC + (size_t)mt * 16 * 256, 256, lane);
    __syncthreads();
    wave_gemm<GMT, 2, 8>(acc, sC, 256, pg3, wave * 2, lane);
    __syncthreads();
#pragma unroll
    for (int t = 0; t < 2; ++t) {
        const int c = wave * 32 + t * 16 + (lane & 15);
        const float b = gb3[c];
#pragma unroll
        for (int mt = 0; mt < GMT; ++mt)
#pragma unroll
            for (int r = 0; r < 8; ++r) {
                int m = mt * 16 + r + half * 8;
                int node = rowBase + m;
                float hv = acc[mt][t][r] + b;
                if (node < nN) hbuf[(size_t)node * 256 + c] = hv;
                sC[(size_t)m * 256 + c] = f2bf(hv);
            }
    }
    __syncthreads();
    // gate: 256 -> 256 -> 256 -> 1
    wave_gemm<GMT, 2, 8>(acc, sC, 256, pt1, wave * 2, lane);
    __syncthreads();
#pragma unroll
    for (int mt = 0; mt < GMT; ++mt)
        wave_gn_relu_store<2>(acc[mt], tg1, tb1, wave * 32,
                              sC + (size_t)mt * 16 * 256, 256, lane);
    __syncthreads();
    wave_gemm<GMT, 2, 8>(acc, sC, 256, pt2, wave * 2, lane);
    __syncthreads();
#pragma unroll
    for (int mt = 0; mt < GMT; ++mt)
        wave_gn_relu_store<2>(acc[mt], tg2, tb2, wave * 32,
                              sC + (size_t)mt * 16 * 256, 256, lane);
    __syncthreads();
    {   // final 256 -> 1 dot per row: 8 segments of 32 channels per row
        int m = tid >> 3, seg = tid & 7;
        float p = 0.f;
#pragma unroll
        for (int j = 0; j < 32; ++j) {
            int k = seg * 32 + j;
            p += bf2f(sC[(size_t)m * 256 + k]) * tw3[k];
        }
        sRed[m * 8 + seg] = p;
    }
    __syncthreads();
    if (tid < GROWS) {
        float s = tb3[0];
#pragma unroll
        for (int j = 0; j < 8; ++j) s += sRed[tid * 8 + j];
        int node = rowBase + tid;
        if (node < nN) gateout[node] = s;
    }
}

// ---------------------------------------------------------------------------
// Attention pooling (batch is sorted -> binary-searched ranges) and decoder.
// ---------------------------------------------------------------------------
__device__ __forceinline__ int lower_bound_i(const int* a, int n, int v) {
    int lo = 0, hi = n;
    while (lo < hi) { int mid = (lo + hi) >> 1; if (a[mid] < v) lo = mid + 1; else hi = mid; }
    return lo;
}

__global__ void gmax_kernel(const float* __restrict__ gate, const int* __restrict__ batch,
                            float* __restrict__ gmax, int n) {
    __shared__ float red[256];
    __shared__ int slo, shi;
    int b = blockIdx.x;
    if (threadIdx.x == 0) { slo = lower_bound_i(batch, n, b); shi = lower_bound_i(batch, n, b + 1); }
    __syncthreads();
    float m = -3.402823e38f;
    for (int i = slo + threadIdx.x; i < shi; i += 256) m = fmaxf(m, gate[i]);
    red[threadIdx.x] = m;
    __syncthreads();
    for (int s = 128; s > 0; s >>= 1) {
        if (threadIdx.x < s) red[threadIdx.x] = fmaxf(red[threadIdx.x], red[threadIdx.x + s]);
        __syncthreads();
    }
    if (threadIdx.x == 0) { float r = red[0]; gmax[b] = (r > -3.0e38f) ? r : 0.f; }
}

__global__ void pool_kernel(const float* __restrict__ gate, const float* __restrict__ h,
                            const int* __restrict__ batch, const float* __restrict__ gmax,
                            float* __restrict__ pooled, int n) {
    __shared__ int slo, shi;
    int b = blockIdx.x, c = threadIdx.x;  // 256 threads = 256 channels
    if (c == 0) { slo = lower_bound_i(batch, n, b); shi = lower_bound_i(batch, n, b + 1); }
    __syncthreads();
    float acc = 0.f, den = 0.f;
    const float gm = gmax[b];
    for (int i = slo; i < shi; ++i) {
        float e = __expf(gate[i] - gm);
        den += e;
        acc += e * h[(size_t)i * 256 + c];
    }
    pooled[(size_t)b * 256 + c] = (den > 0.f) ? acc / den : 0.f;
}

__global__ void dec_kernel(const float* __restrict__ pooled, const float* w1,
                           const float* b1, const float* g1, const float* be1,
                           const float* w2, const float* b2, const float* g2,
                           const float* be2, const float* w3, const float* b3,
                           float* __restrict__ out) {
    __shared__ float sx[256], sh[256], sh2[256];
    int b = blockIdx.x, c = threadIdx.x;
    sx[c] = pooled[(size_t)b * 256 + c];
    __syncthreads();
    float a1 = b1[c];
    for (int k = 0; k < 256; ++k) a1 += sx[k] * w1[k * 256 + c];
    sh[c] = a1;
    __syncthreads();
    {
        int g0 = (c >> 5) << 5;
        float mu = 0.f, qq = 0.f;
        for (int j = 0; j < 32; ++j) { float v = sh[g0 + j]; mu += v; qq += v * v; }
        mu *= (1.f / 32.f);
        float rstd = rsqrtf(qq * (1.f / 32.f) - mu * mu + EPS_GN);
        sh2[c] = fmaxf((a1 - mu) * rstd * g1[c] + be1[c], 0.f);
    }
    __syncthreads();
    float a2 = b2[c];
    for (int k = 0; k < 256; ++k) a2 += sh2[k] * w2[k * 256 + c];
    sh[c] = a2;
    __syncthreads();
    {
        int g0 = (c >> 5) << 5;
        float mu = 0.f, qq = 0.f;
        for (int j = 0; j < 32; ++j) { float v = sh[g0 + j]; mu += v; qq += v * v; }
        mu *= (1.f / 32.f);
        float rstd = rsqrtf(qq * (1.f / 32.f) - mu * mu + EPS_GN);
        sh2[c] = fmaxf((a2 - mu) * rstd * g2[c] + be2[c], 0.f);
    }
    __syncthreads();
    if (c < 6) {
        float o = b3[c];
        for (int k = 0; k < 256; ++k) o += sh2[k] * w3[k * 6 + c];
        out[(size_t)b * 6 + c] = o;
    }
}

// ---------------------------------------------------------------------------
// Host side. Input order = setup_inputs() insertion order, params recursed in
// insertion order (w1,b1,gn1(gamma,beta),w2,b2,gn2(gamma,beta),w3,b3), None
// biases dropped for local/glob/gate blocks.
// ---------------------------------------------------------------------------
enum {
    IN_X = 0, IN_POS = 1,
    ENC_W1 = 2, ENC_B1, ENC_G1, ENC_BE1, ENC_W2, ENC_B2, ENC_G2, ENC_BE2, ENC_W3, ENC_B3,
    EPS_W1 = 12, EPS_B1, EPS_G1, EPS_BE1, EPS_W2, EPS_B2, EPS_G2, EPS_BE2, EPS_W3, EPS_B3,
    LOC_W1 = 22, LOC_G1, LOC_BE1, LOC_W2, LOC_G2, LOC_BE2, LOC_W3, LOC_B3,
    GLB_W1 = 30, GLB_G1, GLB_BE1, GLB_W2, GLB_G2, GLB_BE2, GLB_W3, GLB_B3,
    GAT_W1 = 38, GAT_G1, GAT_BE1, GAT_W2, GAT_G2, GAT_BE2, GAT_W3, GAT_B3,
    DEC_W1 = 46, DEC_B1, DEC_G1, DEC_BE1, DEC_W2, DEC_B2, DEC_G2, DEC_BE2, DEC_W3, DEC_B3,
    IN_EDGE = 56, IN_BATCH = 57, IN_BS = 58
};

extern "C" void kernel_launch(void* const* d_in, const int* in_sizes, int n_in,
                              void* d_out, int out_size, void* d_ws, size_t ws_size,
                              hipStream_t stream) {
    (void)n_in; (void)ws_size;
    auto f32 = [&](int i) { return (const float*)d_in[i]; };

    const float* x   = f32(IN_X);
    const float* pos = f32(IN_POS);
    const int* edge  = (const int*)d_in[IN_EDGE];
    const int* batch = (const int*)d_in[IN_BATCH];

    const int N = in_sizes[IN_X] / 4;
    const int E = in_sizes[IN_EDGE] / 2;
    const int B = out_size / 6;           // q is [B,6]
    const int* src = edge;
    const int* dst = edge + E;

    // ---- workspace carve (256B aligned) ----
    char* wsp = (char*)d_ws;
    auto carve = [&](size_t bytes) -> char* {
        char* p = wsp;
        wsp += (bytes + 255) & ~(size_t)255;
        return p;
    };
    ushort_t* feat  = (ushort_t*)carve((size_t)N * 96 * 2);
    ushort_t* pw_l1 = (ushort_t*)carve((size_t)128 * 512 * 2);
    ushort_t* pw_l2 = (ushort_t*)carve((size_t)512 * 512 * 2);
    ushort_t* pw_l3 = (ushort_t*)carve((size_t)512 * 512 * 2);
    ushort_t* pw_g1 = (ushort_t*)carve((size_t)512 * 256 * 2);
    ushort_t* pw_g2 = (ushort_t*)carve((size_t)256 * 256 * 2);
    ushort_t* pw_g3 = (ushort_t*)carve((size_t)256 * 256 * 2);
    ushort_t* pw_t1 = (ushort_t*)carve((size_t)256 * 256 * 2);
    ushort_t* pw_t2 = (ushort_t*)carve((size_t)256 * 256 * 2);
    float* agg    = (float*)carve((size_t)N * 512 * 4);
    float* hbuf   = (float*)carve((size_t)N * 256 * 4);
    float* gateb  = (float*)carve((size_t)N * 4);
    float* gmax   = (float*)carve((size_t)B * 4);
    float* pooled = (float*)carve((size_t)B * 256 * 4);

    hipMemsetAsync(agg, 0, (size_t)N * 512 * 4, stream);

    // ---- pack weights into WMMA B-fragment order (bf16) ----
    auto pack = [&](const float* W, ushort_t* o, int Kr, int Kp, int No) {
        int total = Kp * No;
        int grid = (total + 255) / 256;
        if (grid > 1024) grid = 1024;
        pack_b_kernel<<<grid, 256, 0, stream>>>(W, o, Kr, Kp, No);
    };
    pack(f32(LOC_W1), pw_l1, 100, 128, 512);
    pack(f32(LOC_W2), pw_l2, 512, 512, 512);
    pack(f32(LOC_W3), pw_l3, 512, 512, 512);
    pack(f32(GLB_W1), pw_g1, 512, 512, 256);
    pack(f32(GLB_W2), pw_g2, 256, 256, 256);
    pack(f32(GLB_W3), pw_g3, 256, 256, 256);
    pack(f32(GAT_W1), pw_t1, 256, 256, 256);
    pack(f32(GAT_W2), pw_t2, 256, 256, 256);

    // ---- encoders -> feat [N,96] bf16 ----
    int encGrid = (N + 255) / 256;
    mlp_node_kernel<4, 32, 4><<<encGrid, 256, 0, stream>>>(
        x, f32(ENC_W1), f32(ENC_B1), f32(ENC_G1), f32(ENC_BE1), f32(ENC_W2),
        f32(ENC_B2), f32(ENC_G2), f32(ENC_BE2), f32(ENC_W3), f32(ENC_B3),
        feat, 96, 0, N);
    mlp_node_kernel<4, 64, 4><<<encGrid, 256, 0, stream>>>(
        pos, f32(EPS_W1), f32(EPS_B1), f32(EPS_G1), f32(EPS_BE1), f32(EPS_W2),
        f32(EPS_B2), f32(EPS_G2), f32(EPS_BE2), f32(EPS_W3), f32(EPS_B3),
        feat, 96, 32, N);

    // ---- fused local message block + scatter-add (the 391 GFLOP kernel) ----
    int locGrid = (E + N + LROWS - 1) / LROWS;
    local_kernel<<<locGrid, 256, 0, stream>>>(
        feat, pos, src, dst, pw_l1, pw_l2, pw_l3, f32(LOC_G1), f32(LOC_BE1),
        f32(LOC_G2), f32(LOC_BE2), f32(LOC_B3), agg, E, N);

    // ---- glob + gate ----
    int ggGrid = (N + GROWS - 1) / GROWS;
    glob_gate_kernel<<<ggGrid, 256, 0, stream>>>(
        agg, pw_g1, pw_g2, pw_g3, f32(GLB_G1), f32(GLB_BE1), f32(GLB_G2),
        f32(GLB_BE2), f32(GLB_B3), pw_t1, pw_t2, f32(GAT_G1), f32(GAT_BE1),
        f32(GAT_G2), f32(GAT_BE2), f32(GAT_W3), f32(GAT_B3), hbuf, gateb, N);

    // ---- attention pooling + decoder ----
    gmax_kernel<<<B, 256, 0, stream>>>(gateb, batch, gmax, N);
    pool_kernel<<<B, 256, 0, stream>>>(gateb, hbuf, batch, gmax, pooled, N);
    dec_kernel<<<B, 256, 0, stream>>>(
        pooled, f32(DEC_W1), f32(DEC_B1), f32(DEC_G1), f32(DEC_BE1), f32(DEC_W2),
        f32(DEC_B2), f32(DEC_G2), f32(DEC_BE2), f32(DEC_W3), f32(DEC_B3),
        (float*)d_out);
}
